// SpatialCrossAttention_15307263443422
// MI455X (gfx1250) — matrix operand order repaired and checked by reading, not compiled, not run on hardware
//
#include <hip/hip_runtime.h>
#include <stdint.h>

// ---------------------------------------------------------------------------
// CDNA5 (gfx1250) wave32 WMMA flash cross-attention
//   q = Wq*water+bq ; k = Wk*bed+bk ; v = Wv*bed+bv   (1x1x1 conv == GEMM)
//   attn = softmax(q^T k / sqrt(c)) ; o = v attn^T ; out = Wo*o + bo
// Layouts chosen for WMMA operand shapes:
//   q    : [n, i, c]  (A operand: lane=row i, contiguous K=c chunks)
//   k    : [n, c, j]  (B operand: K=c per element, lane=N col j; j contiguous)
//   v    : [n, j, c]  (B operand: K=j per element, lane=N col c; c contiguous)
//   obuf : [n, c, i]  (B operand for Wo projection, contiguous N=i elements)
// ---------------------------------------------------------------------------

typedef __attribute__((ext_vector_type(16))) __bf16 v16bf;
typedef __attribute__((ext_vector_type(8)))  float  v8f;
typedef __attribute__((ext_vector_type(4)))  int    v4i;

typedef __attribute__((address_space(1))) v4i  gv4i;   // global int4
typedef __attribute__((address_space(3))) v4i  lv4i;   // LDS int4

static_assert(sizeof(__bf16) == 2, "bf16 must be 2 bytes");

constexpr int C   = 64;
constexpr int Hd  = 48, Wd = 48, T = 8;
constexpr int HW  = Hd * Wd;        // 2304 (sequence length)
constexpr int HWT = HW * T;         // 18432 (voxels per batch b)
constexpr int NB  = 2 * T;          // 16 attention batches (b*t)

#define DEVFN static __device__ __forceinline__

// Native f32 -> bf16 conversion (backend emits v_cvt_pk_bf16_f32 pairs).
DEVFN __bf16 f2bf(float f) { return (__bf16)f; }

DEVFN v8f wmma_bf16(v16bf a, v16bf b, v8f c) {
  // v_wmma_f32_16x16x32_bf16  D = A(16x32) * B(32x16) + C(16x16 f32)
  return __builtin_amdgcn_wmma_f32_16x16x32_bf16(
      /*neg_a=*/false, a, /*neg_b=*/false, b,
      /*c_mod=*/(short)0, c, /*reuse_a=*/false, /*reuse_b=*/false);
}

// 16-bit A-matrix (16x32) K-index owned by element e of lane L (ISA 7.12.2):
//   lanes 0-15 hold K in {0..7, 16..23}; lanes 16-31 hold K in {8..15, 24..31}
//   -> per lane the fragment is two contiguous 16B chunks (compiler merges).
DEVFN int kA(int e, int lane) {
  return ((e < 8) ? e : (e + 8)) + ((lane & 16) ? 8 : 0);
}

// gfx1250 async global->LDS path (ASYNCcnt-tracked, no VGPR round trip)
#if defined(__has_builtin)
#if __has_builtin(__builtin_amdgcn_global_load_async_to_lds_b128) && \
    __has_builtin(__builtin_amdgcn_s_wait_asynccnt)
#define USE_ASYNC_LDS 1
#endif
#endif

// ---------------------------------------------------------------------------
// Kernel 1: fused QKV projection.  3 waves/block: wave0->q, wave1->k, wave2->v
// Per wave: 64 out-channels x 16 voxels via 8 WMMAs.
// ---------------------------------------------------------------------------
__global__ __launch_bounds__(96)
void qkv_proj_kernel(const float* __restrict__ water, const float* __restrict__ bed,
                     const float* __restrict__ Wq, const float* __restrict__ bq,
                     const float* __restrict__ Wk, const float* __restrict__ bk,
                     const float* __restrict__ Wv, const float* __restrict__ bv,
                     __bf16* __restrict__ q, __bf16* __restrict__ k,
                     __bf16* __restrict__ v) {
  const int wave = threadIdx.x >> 5;
  const int lane = threadIdx.x & 31;
  const int b    = blockIdx.y;
  const int p0   = blockIdx.x * 16;           // voxel tile within batch b

  const float* X    = (wave == 0) ? water : bed;
  const float* Wm   = (wave == 0) ? Wq : (wave == 1) ? Wk : Wv;
  const float* bias = (wave == 0) ? bq : (wave == 1) ? bk : bv;
  __bf16*      dst  = (wave == 0) ? q  : (wave == 1) ? k  : v;

  // B fragments: K row = input channel kA(e, lane) (element), N col = voxel (lane & 15)
  v16bf Bf[2];
#pragma unroll
  for (int ks = 0; ks < 2; ++ks) {
    const float* xp = X + ((size_t)b * C + ks * 32) * HWT + p0 + (lane & 15);
#pragma unroll
    for (int e = 0; e < 16; ++e) Bf[ks][e] = f2bf(xp[(size_t)kA(e, lane) * HWT]);
  }

  const v8f vzero = {};
  v8f acc[4];
#pragma unroll
  for (int mt = 0; mt < 4; ++mt) acc[mt] = vzero;

#pragma unroll
  for (int ks = 0; ks < 2; ++ks) {            // K (input channel) steps of 32
#pragma unroll
    for (int mt = 0; mt < 4; ++mt) {          // output-channel tiles share B
      v16bf Af;
      const float* wr = Wm + (size_t)(mt * 16 + (lane & 15)) * C + ks * 32;
#pragma unroll
      for (int e = 0; e < 16; ++e) Af[e] = f2bf(wr[kA(e, lane)]);
      acc[mt] = wmma_bf16(Af, Bf[ks], acc[mt]);
    }
  }

  const int p  = p0 + (lane & 15);            // this lane's voxel (D column)
  const int tt = p & (T - 1);
  const int i  = p >> 3;                      // spatial index (t innermost)
  const int n  = b * T + tt;

  if (wave == 1) {
    // k stored transposed [n, c, j] for vectorized B-fragment loads in attention
#pragma unroll
    for (int mt = 0; mt < 4; ++mt) {
#pragma unroll
      for (int r = 0; r < 8; ++r) {
        const int oc = mt * 16 + r + ((lane & 16) ? 8 : 0);
        dst[((size_t)n * C + oc) * HW + i] = f2bf(acc[mt][r] + bias[oc]);
      }
    }
  } else {
    // q and v stored [n, i, c] (contiguous 8-element runs per lane)
    __bf16* drow = dst + ((size_t)n * HW + i) * C;
#pragma unroll
    for (int mt = 0; mt < 4; ++mt) {
#pragma unroll
      for (int r = 0; r < 8; ++r) {
        const int oc = mt * 16 + r + ((lane & 16) ? 8 : 0);
        drow[oc] = f2bf(acc[mt][r] + bias[oc]);
      }
    }
  }
}

// ---------------------------------------------------------------------------
// Kernel 2: flash attention. Block = 4 waves = 64 query rows of one batch n.
// Streams K/V in 64-row blocks through LDS (async-to-LDS when available);
// online softmax; PV accumulated in f32 WMMA accumulators.
// ---------------------------------------------------------------------------
__global__ __launch_bounds__(128)
void flash_attn_kernel(const __bf16* __restrict__ q, const __bf16* __restrict__ kmat,
                       const __bf16* __restrict__ vmat, __bf16* __restrict__ obuf) {
  __shared__ __attribute__((aligned(32))) __bf16 Klds[64 * 64];   // [c][j]  8 KB
  __shared__ __attribute__((aligned(32))) __bf16 Vlds[64 * 64];   // [j][c]  8 KB
  __shared__ __attribute__((aligned(32))) __bf16 Plds[4][16 * 64];// 8 KB (wave-private)

  const int wave = threadIdx.x >> 5;
  const int lane = threadIdx.x & 31;
  const int n    = blockIdx.y;
  const int i0   = blockIdx.x * 64;
  const int irow = i0 + wave * 16 + (lane & 15);   // A-fragment row for this lane

  // Q A-fragments (reused across the whole j loop)
  v16bf Qa[2];
  const __bf16* qrow = q + ((size_t)n * HW + irow) * C;
#pragma unroll
  for (int ks = 0; ks < 2; ++ks)
#pragma unroll
    for (int e = 0; e < 16; ++e) Qa[ks][e] = qrow[ks * 32 + kA(e, lane)];

  // Online-softmax state: per C/D-layout row r (lane half selects r vs r+8)
  float m[8], l[8];
  v8f  O[4];
  const v8f vzero = {};
#pragma unroll
  for (int r = 0; r < 8; ++r) { m[r] = -1.0e30f; l[r] = 0.0f; }
#pragma unroll
  for (int ct = 0; ct < 4; ++ct) O[ct] = vzero;

  const float sc = 0.125f * 1.44269504088896f;  // (1/sqrt(64)) * log2(e)
  const int NJB = HW / 64;                      // 36 key/value blocks

  const __bf16* gkt = kmat + (size_t)n * C * HW;        // k rows: [c][j]
  for (int jb = 0; jb < NJB; ++jb) {
    const int j0 = jb * 64;
    __syncthreads();   // previous iteration finished reading LDS
    {
      const __bf16* gv = vmat + ((size_t)n * HW + j0) * C;
#if defined(USE_ASYNC_LDS)
#pragma unroll
      for (int it = 0; it < 4; ++it) {
        const int cidx = it * 128 + threadIdx.x;        // 16B chunk id 0..511
        const int krow = cidx >> 3;                     // c row
        const int kcol = (cidx & 7) * 8;                // bf16 offset in row
        __builtin_amdgcn_global_load_async_to_lds_b128(
            (gv4i*)(gkt + (size_t)krow * HW + j0 + kcol),
            (lv4i*)(Klds + krow * 64 + kcol), 0, 0);
        __builtin_amdgcn_global_load_async_to_lds_b128(
            (gv4i*)(gv + cidx * 8), (lv4i*)(Vlds + cidx * 8), 0, 0);
      }
      __builtin_amdgcn_s_wait_asynccnt(0);
#else
#pragma unroll
      for (int it = 0; it < 4; ++it) {
        const int cidx = it * 128 + threadIdx.x;
        const int krow = cidx >> 3;
        const int kcol = (cidx & 7) * 8;
        *(uint4*)(Klds + krow * 64 + kcol) =
            *(const uint4*)(gkt + (size_t)krow * HW + j0 + kcol);
        *(uint4*)(Vlds + cidx * 8) = *(const uint4*)(gv + cidx * 8);
      }
#endif
    }
    if (jb + 1 < NJB) {  // global_prefetch_b8 of next block (per-lane rows)
      __builtin_prefetch(gkt + (size_t)lane * 2 * HW + j0 + 64, 0, 0);
      __builtin_prefetch(vmat + ((size_t)n * HW + j0 + 64) * C + lane * 128, 0, 0);
    }
    __syncthreads();

    // S = Q * K^T : B fragments gather Klds rows kA(e, lane), column lane & 15
    v8f S[4];
#pragma unroll
    for (int nt = 0; nt < 4; ++nt) S[nt] = vzero;
#pragma unroll
    for (int ks = 0; ks < 2; ++ks) {
#pragma unroll
      for (int nt = 0; nt < 4; ++nt) {
        v16bf Bf;
#pragma unroll
        for (int e = 0; e < 16; ++e)
          Bf[e] = Klds[(ks * 32 + kA(e, lane)) * 64 + nt * 16 + (lane & 15)];
        S[nt] = wmma_bf16(Qa[ks], Bf, S[nt]);
      }
    }

    // Online softmax. C-layout: lane = column j, VGPR r = row (r / r+8 by half).
    float f[8];
#pragma unroll
    for (int r = 0; r < 8; ++r) {
      float x = fmaxf(fmaxf(S[0][r], S[1][r]), fmaxf(S[2][r], S[3][r])) * sc;
      x = fmaxf(x, __shfl_xor(x, 1));
      x = fmaxf(x, __shfl_xor(x, 2));
      x = fmaxf(x, __shfl_xor(x, 4));
      x = fmaxf(x, __shfl_xor(x, 8));    // reduce across the 16-lane half
      const float mn = fmaxf(m[r], x);
      f[r] = exp2f(m[r] - mn);
      m[r] = mn;
    }

    float rs[8];
#pragma unroll
    for (int r = 0; r < 8; ++r) rs[r] = 0.0f;
#pragma unroll
    for (int nt = 0; nt < 4; ++nt) {
#pragma unroll
      for (int r = 0; r < 8; ++r) {
        const float p = exp2f(S[nt][r] * sc - m[r]);
        rs[r] += p;
        // stage P row-major in wave-private LDS (C-layout -> A-layout transpose)
        Plds[wave][(r + ((lane & 16) ? 8 : 0)) * 64 + nt * 16 + (lane & 15)] = f2bf(p);
      }
    }
#pragma unroll
    for (int r = 0; r < 8; ++r) {
      float s = rs[r];
      s += __shfl_xor(s, 1);
      s += __shfl_xor(s, 2);
      s += __shfl_xor(s, 4);
      s += __shfl_xor(s, 8);
      l[r] = l[r] * f[r] + s;
#pragma unroll
      for (int ct = 0; ct < 4; ++ct) O[ct][r] *= f[r];
    }

    // O += P * V : A (=P) held constant across the 4 c-tiles
#pragma unroll
    for (int js = 0; js < 2; ++js) {
      v16bf Pa;
#pragma unroll
      for (int e = 0; e < 16; ++e)
        Pa[e] = Plds[wave][(lane & 15) * 64 + js * 32 + kA(e, lane)];
#pragma unroll
      for (int ct = 0; ct < 4; ++ct) {
        v16bf Vf;
#pragma unroll
        for (int e = 0; e < 16; ++e)
          Vf[e] = Vlds[(js * 32 + kA(e, lane)) * 64 + ct * 16 + (lane & 15)];
        O[ct] = wmma_bf16(Pa, Vf, O[ct]);
      }
    }
  }

  // Normalize and write O transposed [n, c, i]: 8-element contiguous runs/lane,
  // and kernel 3's B-fragment loads are contiguous across lanes (i).
#pragma unroll
  for (int ct = 0; ct < 4; ++ct) {
    const int cc = ct * 16 + (lane & 15);
    __bf16* op = obuf + ((size_t)n * C + cc) * HW + i0 + wave * 16 +
                 ((lane & 16) ? 8 : 0);
#pragma unroll
    for (int r = 0; r < 8; ++r) op[r] = f2bf(O[ct][r] / l[r]);
  }
}

// ---------------------------------------------------------------------------
// Kernel 3: output projection Wo * O + bo, scatter back to [b, c, h, w, t] f32.
// 4 waves/block, wave = output-channel tile; 16 sequence positions per block.
// ---------------------------------------------------------------------------
__global__ __launch_bounds__(128)
void out_proj_kernel(const __bf16* __restrict__ obuf, const float* __restrict__ Wo,
                     const float* __restrict__ bo, float* __restrict__ out) {
  const int wave = threadIdx.x >> 5;          // M tile (output-channel block)
  const int lane = threadIdx.x & 31;
  const int n    = blockIdx.y;
  const int i0   = blockIdx.x * 16;

  // B fragments: K row = channel kA(e, lane) (element), N col = position (lane & 15)
  v16bf Bf[2];
#pragma unroll
  for (int ks = 0; ks < 2; ++ks)
#pragma unroll
    for (int e = 0; e < 16; ++e)
      Bf[ks][e] = obuf[((size_t)n * C + ks * 32 + kA(e, lane)) * HW + i0 + (lane & 15)];

  v8f acc = {};
#pragma unroll
  for (int ks = 0; ks < 2; ++ks) {
    v16bf Af;
    const float* wr = Wo + (size_t)(wave * 16 + (lane & 15)) * C + ks * 32;
#pragma unroll
    for (int e = 0; e < 16; ++e) Af[e] = f2bf(wr[kA(e, lane)]);
    acc = wmma_bf16(Af, Bf[ks], acc);
  }

  const int b  = n >> 3;
  const int tt = n & (T - 1);
  const int i  = i0 + (lane & 15);            // this lane's D column
#pragma unroll
  for (int r = 0; r < 8; ++r) {
    const int oc = wave * 16 + r + ((lane & 16) ? 8 : 0);
    out[(((size_t)b * C + oc) * HW + i) * T + tt] = acc[r] + bo[oc];
  }
}

// ---------------------------------------------------------------------------
extern "C" void kernel_launch(void* const* d_in, const int* in_sizes, int n_in,
                              void* d_out, int out_size, void* d_ws, size_t ws_size,
                              hipStream_t stream) {
  const float* water = (const float*)d_in[0];
  const float* bed   = (const float*)d_in[1];
  const float* Wq    = (const float*)d_in[2];
  const float* bq    = (const float*)d_in[3];
  const float* Wk    = (const float*)d_in[4];
  const float* bk    = (const float*)d_in[5];
  const float* Wv    = (const float*)d_in[6];
  const float* bv    = (const float*)d_in[7];
  const float* Wo    = (const float*)d_in[8];
  const float* bo    = (const float*)d_in[9];
  float* out = (float*)d_out;

  // workspace: bf16 q, k, v, o  each NB*HW*C elements (4 * 4.72 MB = ~18.9 MB)
  const size_t QSZ = (size_t)NB * HW * C;
  __bf16* q  = (__bf16*)d_ws;
  __bf16* k  = q + QSZ;
  __bf16* v  = k + QSZ;
  __bf16* ob = v + QSZ;

  qkv_proj_kernel<<<dim3(HWT / 16, 2), 96, 0, stream>>>(
      water, bed, Wq, bq, Wk, bk, Wv, bv, q, k, v);
  flash_attn_kernel<<<dim3(HW / 64, NB), 128, 0, stream>>>(q, k, v, ob);
  out_proj_kernel<<<dim3(HW / 16, NB), 128, 0, stream>>>(ob, Wo, bo, out);
}
